// RGCN_3006477107337
// MI455X (gfx1250) — compile-verified, matrix operand order checked by reading, not run on hardware
//
#include <hip/hip_runtime.h>
#include <hip/hip_bf16.h>
#include <stdint.h>

#define N_NODES 100000
#define N_EDGES 1600000
#define DIM 128

typedef __attribute__((ext_vector_type(16))) __bf16 bf16x16;
typedef __attribute__((ext_vector_type(8)))  float  f32x8;

union BfPack {
    unsigned int u[8];
    bf16x16 v;
};

__device__ __forceinline__ unsigned int pack_bf16(float lo, float hi) {
    unsigned int ul = __float_as_uint(lo);
    unsigned int uh = __float_as_uint(hi);
    // round-to-nearest-even bf16
    ul = (ul + 0x7FFFu + ((ul >> 16) & 1u)) >> 16;
    uh = (uh + 0x7FFFu + ((uh >> 16) & 1u)) >> 16;
    return (uh << 16) | (ul & 0xFFFFu);
}

// ---------------------------------------------------------------------------
// Weight prep: W1t[r][o][i] = bf16( sum_b comb1[r,b] * basis1[b,i,o] )
// stored transposed (output-major, K contiguous) to match WMMA B-frag loads.
// ---------------------------------------------------------------------------
__global__ void build_w1t(const float* __restrict__ basis1,
                          const float* __restrict__ comb1,
                          unsigned short* __restrict__ w1t) {
    int idx = blockIdx.x * blockDim.x + threadIdx.x;     // 8*128*128
    if (idx >= 8 * DIM * DIM) return;
    int r = idx >> 14;
    int n = (idx >> 7) & 127;   // output col o
    int k = idx & 127;          // input  row i
    float s = 0.f;
#pragma unroll
    for (int b = 0; b < 8; ++b)
        s += comb1[r * 8 + b] * basis1[((size_t)(b * DIM + k)) * DIM + n];
    unsigned int u = __float_as_uint(s);
    w1t[idx] = (unsigned short)((u + 0x7FFFu + ((u >> 16) & 1u)) >> 16);
}

__global__ void build_loop1t(const float* __restrict__ loop1,
                             unsigned short* __restrict__ l1t) {
    int idx = blockIdx.x * blockDim.x + threadIdx.x;     // 128*128
    if (idx >= DIM * DIM) return;
    int n = idx >> 7;           // output col
    int k = idx & 127;          // input row
    float s = loop1[(size_t)k * DIM + n];
    unsigned int u = __float_as_uint(s);
    l1t[idx] = (unsigned short)((u + 0x7FFFu + ((u >> 16) & 1u)) >> 16);
}

// w2col[r][i] = sum_b comb2[r,b] * basis2[b,i,0]
__global__ void build_w2col(const float* __restrict__ basis2,
                            const float* __restrict__ comb2,
                            float* __restrict__ w2col) {
    int idx = blockIdx.x * blockDim.x + threadIdx.x;     // 8*128
    if (idx >= 8 * DIM) return;
    int r = idx >> 7;
    int k = idx & 127;
    float s = 0.f;
#pragma unroll
    for (int b = 0; b < 8; ++b)
        s += comb2[r * 8 + b] * basis2[b * DIM + k];
    w2col[idx] = s;
}

// ---------------------------------------------------------------------------
// Elementwise helpers
// ---------------------------------------------------------------------------
__global__ void init_bias(float* __restrict__ agg, const float* __restrict__ bias1) {
    int i = blockIdx.x * blockDim.x + threadIdx.x;
    if (i < N_NODES * DIM) agg[i] = bias1[i & 127];
}

__global__ void zero_buf(float* __restrict__ p) {
    int i = blockIdx.x * blockDim.x + threadIdx.x;
    if (i < N_NODES * DIM) p[i] = 0.f;
}

__global__ void relu_ip(float* __restrict__ p) {
    int i = blockIdx.x * blockDim.x + threadIdx.x;
    if (i < N_NODES * DIM) p[i] = fmaxf(p[i], 0.f);
}

// ---------------------------------------------------------------------------
// Layer-1 per-relation edge scatter: one wave32 per edge, lane = 4 floats.
// S[dst] += x[src] for edges of relation `rel`. x/S are L2-resident (51 MB
// each, 192 MB L2), so the 1.6 GB of edge feature traffic stays on-chip.
// ---------------------------------------------------------------------------
__global__ __launch_bounds__(256) void scatter_rel(const float* __restrict__ x,
                                                   const int* __restrict__ src,
                                                   const int* __restrict__ dst,
                                                   const int* __restrict__ et,
                                                   float* __restrict__ S, int rel) {
    int wid  = (blockIdx.x * blockDim.x + threadIdx.x) >> 5;  // edge id
    int lane = threadIdx.x & 31;
    if (wid >= N_EDGES) return;
    if (et[wid] != rel) return;
    int s = src[wid], d = dst[wid];
    float4 v = ((const float4*)(x + (size_t)s * DIM))[lane];
    float* o = S + (size_t)d * DIM + lane * 4;
    atomicAdd(o + 0, v.x);
    atomicAdd(o + 1, v.y);
    atomicAdd(o + 2, v.z);
    atomicAdd(o + 3, v.w);
}

// ---------------------------------------------------------------------------
// WMMA GEMM: C(M x128) += A(M x128, fp32) @ Bt^T, Bt packed bf16 [n][k].
// Block = 256 threads = 8 waves; tile 16 rows x 128 cols; wave w -> cols 16w.
//
// Phase 1 (once per block): cooperative coalesced load of the fp32 A tile,
// fp32 -> packed-bf16 conversion in registers (4 packs/thread), store packed
// tile to LDS with a 68-dword padded row stride (68 mod 64 == 4, so each
// ds_load_b128 fragment read touches every bank exactly twice -> no hotspot).
//
// Phase 2 (per wave, K=128): per kstep just 2x ds_load_b128 (A frag, already
// in the ISA 16-bit A-matrix bit layout), 2x b128 global load (B frag from
// pre-transposed weights, L0-hot), 1x v_wmma_f32_16x16x32_bf16. No VALU
// conversion in the hot loop.
// ---------------------------------------------------------------------------
__global__ __launch_bounds__(256) void gemm_tile16(const float* __restrict__ A,
                                                   const unsigned int* __restrict__ Bt,
                                                   float* __restrict__ C) {
    __shared__ unsigned int Abf[16 * 68];   // bf16-pair dwords, padded stride
    const int row0 = blockIdx.x * 16;
    const int t = threadIdx.x;
    {   // phase 1: load + convert 16x128 fp32 tile (8 floats / thread)
        const int m = t >> 4;
        const int k = (t & 15) * 8;
        const float4* g = (const float4*)(A + (size_t)(row0 + m) * DIM + k);
        float4 a0 = g[0], a1 = g[1];
        uint4 p;
        p.x = pack_bf16(a0.x, a0.y);
        p.y = pack_bf16(a0.z, a0.w);
        p.z = pack_bf16(a1.x, a1.y);
        p.w = pack_bf16(a1.z, a1.w);
        *(uint4*)(Abf + m * 68 + (k >> 1)) = p;
    }
    __syncthreads();

    const int lane = t & 31;
    const int n0   = (t >> 5) * 16;           // wave's column block
    const int ncol = n0 + (lane & 15);
    // lane-private fragment base addresses (khalf folded in as +4 dwords)
    const unsigned int* arow = Abf + (lane & 15) * 68 + ((lane >> 4) << 2);
    const unsigned int* brow = Bt + (size_t)ncol * 64 + ((lane >> 4) << 2);

    f32x8 c = {0.f, 0.f, 0.f, 0.f, 0.f, 0.f, 0.f, 0.f};

#pragma unroll
    for (int kstep = 0; kstep < 4; ++kstep) {
        BfPack a, b;
        uint4 a0 = *(const uint4*)(arow + kstep * 16);       // K = k0+khalf..+7
        uint4 a1 = *(const uint4*)(arow + kstep * 16 + 8);   // K = +16..+23
        uint4 b0 = *(const uint4*)(brow + kstep * 16);
        uint4 b1 = *(const uint4*)(brow + kstep * 16 + 8);
        a.u[0] = a0.x; a.u[1] = a0.y; a.u[2] = a0.z; a.u[3] = a0.w;
        a.u[4] = a1.x; a.u[5] = a1.y; a.u[6] = a1.z; a.u[7] = a1.w;
        b.u[0] = b0.x; b.u[1] = b0.y; b.u[2] = b0.z; b.u[3] = b0.w;
        b.u[4] = b1.x; b.u[5] = b1.y; b.u[6] = b1.z; b.u[7] = b1.w;

        c = __builtin_amdgcn_wmma_f32_16x16x32_bf16(false, a.v, false, b.v,
                                                    (short)0, c, false, false);
    }

    // C layout: VGPR j -> row j (lanes 0-15) / j+8 (lanes 16-31), col = lane&15
#pragma unroll
    for (int j = 0; j < 8; ++j) {
        int row = row0 + j + ((lane >> 4) << 3);
        C[(size_t)row * DIM + ncol] += c[j];
    }
}

// ---------------------------------------------------------------------------
// Layer 2: one wave per node. 9 dot-128 products (8 relation cols + loop2),
// wave32 shuffle reduction. out[n] = h1.loop2 + bias2; proj2[r][n] = h1.w2col[r]
// ---------------------------------------------------------------------------
__global__ __launch_bounds__(256) void layer2_proj(const float* __restrict__ h1,
                                                   const float* __restrict__ w2col,
                                                   const float* __restrict__ loop2,
                                                   const float* __restrict__ bias2,
                                                   float* __restrict__ out,
                                                   float* __restrict__ proj2) {
    int n    = (blockIdx.x * blockDim.x + threadIdx.x) >> 5;
    int lane = threadIdx.x & 31;
    if (n >= N_NODES) return;
    float4 h4 = ((const float4*)(h1 + (size_t)n * DIM))[lane];
    float acc[9];
#pragma unroll
    for (int r = 0; r < 8; ++r) {
        float4 w4 = ((const float4*)(w2col + r * DIM))[lane];
        acc[r] = h4.x * w4.x + h4.y * w4.y + h4.z * w4.z + h4.w * w4.w;
    }
    {
        float4 w4 = ((const float4*)loop2)[lane];
        acc[8] = h4.x * w4.x + h4.y * w4.y + h4.z * w4.z + h4.w * w4.w;
    }
#pragma unroll
    for (int v = 0; v < 9; ++v)
        for (int o = 16; o > 0; o >>= 1)
            acc[v] += __shfl_xor(acc[v], o, 32);
    if (lane == 0) {
        out[n] = acc[8] + bias2[0];
#pragma unroll
        for (int r = 0; r < 8; ++r)
            proj2[(size_t)r * N_NODES + n] = acc[r];
    }
}

__global__ void edge2(const int* __restrict__ src, const int* __restrict__ dst,
                      const int* __restrict__ et, const float* __restrict__ proj2,
                      float* __restrict__ out) {
    int e = blockIdx.x * blockDim.x + threadIdx.x;
    if (e >= N_EDGES) return;
    atomicAdd(&out[dst[e]], proj2[(size_t)et[e] * N_NODES + src[e]]);
}

// ---------------------------------------------------------------------------
extern "C" void kernel_launch(void* const* d_in, const int* in_sizes, int n_in,
                              void* d_out, int out_size, void* d_ws, size_t ws_size,
                              hipStream_t stream) {
    const float* x      = (const float*)d_in[0];
    const int*   src    = (const int*)d_in[1];
    const int*   dst    = (const int*)d_in[2];
    const int*   et     = (const int*)d_in[3];
    const float* basis1 = (const float*)d_in[4];
    const float* comb1  = (const float*)d_in[5];
    const float* loop1  = (const float*)d_in[6];
    const float* bias1  = (const float*)d_in[7];
    const float* basis2 = (const float*)d_in[8];
    const float* comb2  = (const float*)d_in[9];
    const float* loop2  = (const float*)d_in[10];
    const float* bias2  = (const float*)d_in[11];
    float* out = (float*)d_out;

    // workspace layout (~106 MB, 16B-aligned offsets)
    char* ws = (char*)d_ws;
    float*          agg   = (float*)(ws);                          // 51,200,000 B
    float*          S     = (float*)(ws + 51200000);               // 51,200,000 B
    unsigned short* W1t   = (unsigned short*)(ws + 102400000);     //    262,144 B
    unsigned short* L1t   = (unsigned short*)(ws + 102662144);     //     32,768 B
    float*          w2col = (float*)(ws + 102694912);              //      4,096 B
    float*          proj2 = (float*)(ws + 102699008);              //  3,200,000 B

    // weight prep
    build_w1t<<<512, 256, 0, stream>>>(basis1, comb1, W1t);
    build_loop1t<<<64, 256, 0, stream>>>(loop1, L1t);
    build_w2col<<<4, 256, 0, stream>>>(basis2, comb2, w2col);

    // layer 1: agg = bias1 + sum_r (scatter_r(x)) @ W1_r + x @ loop1
    init_bias<<<50000, 256, 0, stream>>>(agg, bias1);
    for (int r = 0; r < 8; ++r) {
        zero_buf<<<50000, 256, 0, stream>>>(S);
        scatter_rel<<<200000, 256, 0, stream>>>(x, src, dst, et, S, r);
        gemm_tile16<<<6250, 256, 0, stream>>>(
            S, (const unsigned int*)W1t + (size_t)r * 8192, agg);
    }
    gemm_tile16<<<6250, 256, 0, stream>>>(x, (const unsigned int*)L1t, agg);
    relu_ip<<<50000, 256, 0, stream>>>(agg);   // agg now = h1

    // layer 2
    layer2_proj<<<12500, 256, 0, stream>>>(agg, w2col, loop2, bias2, out, proj2);
    edge2<<<6250, 256, 0, stream>>>(src, dst, et, proj2, out);
}
